// GraphUNet_13099650253564
// MI455X (gfx1250) — matrix operand order repaired and checked by reading, not compiled, not run on hardware
//
#include <hip/hip_runtime.h>
#include <hip/hip_bf16.h>

// ---------------------------------------------------------------------------
// Graph U-Net forward for MI455X (gfx1250, wave32, WMMA).
// Heavy math (Ma@Ma, MnT@H, x@W) runs through one WMMA f16->f32 tiled GEMM
// with double-buffered LDS and vectorized (b128) staging where aligned.
// ---------------------------------------------------------------------------

typedef __attribute__((ext_vector_type(16))) _Float16 v16h;
typedef __attribute__((ext_vector_type(8)))  float    v8f;

#define TM 64
#define TN 64
#define TK 32
#define KPAD 8   // LDS row padding (f16 elems) to spread banks

// ---------------- elementwise / graph-prep kernels -------------------------

__global__ __launch_bounds__(256) void fill_f32(float* p, float v, int count) {
  int i = blockIdx.x * blockDim.x + threadIdx.x;
  if (i < count) p[i] = v;
}

__global__ __launch_bounds__(256) void copy_f32(float* __restrict__ dst,
                                                const float* __restrict__ src, int count) {
  int i = blockIdx.x * blockDim.x + threadIdx.x;
  if (i < count) dst[i] = src[i];
}

// M[src, dst] += 1 for each edge (integer-valued fp adds -> deterministic).
__global__ __launch_bounds__(256) void build_adj(const int* __restrict__ ei,
                                                 float* __restrict__ M, int E, int n) {
  int e = blockIdx.x * blockDim.x + threadIdx.x;
  if (e < E) {
    int s = ei[e];
    int d = ei[E + e];
    atomicAdd(&M[(size_t)s * n + d], 1.0f);
  }
}

__global__ __launch_bounds__(256) void clear_diag(float* __restrict__ M, int n) {
  int i = blockIdx.x * blockDim.x + threadIdx.x;
  if (i < n) M[(size_t)i * n + i] = 0.0f;
}

// Ma = M with diagonal forced to 1 (remove self loops, add w=1 self loops).
__global__ __launch_bounds__(256) void prep_aug(const float* __restrict__ M,
                                                float* __restrict__ Ma, int n) {
  int t = blockIdx.x * blockDim.x + threadIdx.x;
  if (t < n * n) {
    int i = t / n, j = t - i * n;
    Ma[t] = (i == j) ? 1.0f : M[t];
  }
}

// dinv[j] = rsqrt( sum_i M[i,j] + 2 )   (improved GCN: self loop weight 2)
__global__ __launch_bounds__(256) void col_degree_dinv(const float* __restrict__ M,
                                                       float* __restrict__ dinv, int n) {
  int j = blockIdx.x * blockDim.x + threadIdx.x;
  if (j < n) {
    float s = 2.0f;
    for (int i = 0; i < n; ++i) s += M[(size_t)i * n + j];
    dinv[j] = rsqrtf(s);
  }
}

// MnT[i,j] = dinv[i]*dinv[j]*M[j,i] + (i==j ? 2*dinv[i]^2 : 0)
__global__ __launch_bounds__(256) void make_mnT(const float* __restrict__ M,
                                                const float* __restrict__ dinv,
                                                float* __restrict__ out, int n) {
  int t = blockIdx.x * blockDim.x + threadIdx.x;
  if (t < n * n) {
    int i = t / n, j = t - i * n;
    float v = dinv[i] * dinv[j] * M[(size_t)j * n + i];
    if (i == j) v += 2.0f * dinv[i] * dinv[i];
    out[t] = v;
  }
}

// ---------------- WMMA GEMM: C[M,N] = A[M,K] @ B[K,N] (+bias)(+relu) -------
// flags: bit0 = add bias[col], bit1 = relu
__global__ __launch_bounds__(128) void gemm_wmma(const float* __restrict__ A,
                                                 const float* __restrict__ B,
                                                 float* __restrict__ C,
                                                 const float* __restrict__ bias,
                                                 int M, int N, int K, int flags) {
  __shared__ _Float16 As[2][TM][TK + KPAD];     // row-major M x K tiles (ping-pong)
  __shared__ _Float16 Bs[2][TN][TK + KPAD];     // B tiles stored N-major (transposed)

  const int tid  = threadIdx.x;
  const int lane = tid & 31;
  const int wave = tid >> 5;
  const int wm   = wave >> 1;                   // wave row (0..1) -> 32 rows
  const int wn   = wave & 1;                    // wave col (0..1) -> 32 cols
  const int bm   = blockIdx.y * TM;
  const int bn   = blockIdx.x * TN;

  const int r = lane & 15;                      // fragment row/col within 16
  const int h = lane >> 4;                      // lane half

  const bool fullMN = (bm + TM <= M) && (bn + TN <= N);
  const bool vecA = ((K & 3) == 0) && ((((size_t)A) & 15) == 0);
  const bool vecB = ((N & 3) == 0) && ((((size_t)B) & 15) == 0);

  v8f acc[2][2];
  v8f zf = {};
#pragma unroll
  for (int mi = 0; mi < 2; ++mi)
#pragma unroll
    for (int ni = 0; ni < 2; ++ni) acc[mi][ni] = zf;

  float va[16], vb[16];

  // ---- issue global loads for the k-tile starting at k0 into va/vb --------
  auto load_tiles = [&](int k0) {
    const bool fullK = (k0 + TK <= K);
    if (fullMN && fullK) {
      if (vecA) {
        const float4* A4 = (const float4*)A;
        const int Kv = K >> 2;
#pragma unroll
        for (int s = 0; s < 4; ++s) {
          int f = tid + 128 * s;                // 512 float4: row=f/8, kv=f%8
          float4 q = A4[(size_t)(bm + (f >> 3)) * Kv + ((k0 >> 2) + (f & 7))];
          va[4 * s + 0] = q.x; va[4 * s + 1] = q.y;
          va[4 * s + 2] = q.z; va[4 * s + 3] = q.w;
        }
      } else {
#pragma unroll
        for (int s = 0; s < 16; ++s) {
          int idx = tid + 128 * s;              // row = idx/32, kk = idx%32
          va[s] = A[(size_t)(bm + (idx >> 5)) * K + (k0 + (idx & 31))];
        }
      }
      if (vecB) {
        const float4* B4 = (const float4*)B;
        const int Nv = N >> 2;
#pragma unroll
        for (int s = 0; s < 4; ++s) {
          int f = tid + 128 * s;                // 512 float4: kk=f/16, nn4=f%16
          float4 q = B4[(size_t)(k0 + (f >> 4)) * Nv + ((bn >> 2) + (f & 15))];
          vb[4 * s + 0] = q.x; vb[4 * s + 1] = q.y;
          vb[4 * s + 2] = q.z; vb[4 * s + 3] = q.w;
        }
      } else {
#pragma unroll
        for (int s = 0; s < 16; ++s) {
          int idx = tid + 128 * s;              // kk = idx/64, nn = idx%64
          vb[s] = B[(size_t)(k0 + (idx >> 6)) * N + (bn + (idx & 63))];
        }
      }
    } else {
      // edge path: clamp addresses (always legal) + select 0 -> branch-free
#pragma unroll
      for (int s = 0; s < 16; ++s) {
        int idx = tid + 128 * s;
        int row = bm + (idx >> 5);
        int gk  = k0 + (idx & 31);
        int rc  = row < M ? row : M - 1;
        int kc  = gk < K ? gk : K - 1;
        float v = A[(size_t)rc * K + kc];
        va[s] = (row < M && gk < K) ? v : 0.0f;
      }
#pragma unroll
      for (int s = 0; s < 16; ++s) {
        int idx = tid + 128 * s;
        int gk  = k0 + (idx >> 6);
        int col = bn + (idx & 63);
        int kc  = gk < K ? gk : K - 1;
        int cc  = col < N ? col : N - 1;
        float v = B[(size_t)kc * N + cc];
        vb[s] = (gk < K && col < N) ? v : 0.0f;
      }
    }
  };

  // ---- convert + store va/vb into LDS buffer `buf` ------------------------
  auto store_tiles = [&](int buf, bool edge) {
    if (!edge && vecA) {
#pragma unroll
      for (int s = 0; s < 4; ++s) {
        int f = tid + 128 * s;
        int row = f >> 3, kk = (f & 7) << 2;
#pragma unroll
        for (int c = 0; c < 4; ++c)
          As[buf][row][kk + c] = (_Float16)va[4 * s + c];
      }
    } else {
#pragma unroll
      for (int s = 0; s < 16; ++s) {
        int idx = tid + 128 * s;
        As[buf][idx >> 5][idx & 31] = (_Float16)va[s];
      }
    }
    if (!edge && vecB) {
#pragma unroll
      for (int s = 0; s < 4; ++s) {
        int f = tid + 128 * s;
        int kk = f >> 4, nn = (f & 15) << 2;
#pragma unroll
        for (int c = 0; c < 4; ++c)
          Bs[buf][nn + c][kk] = (_Float16)vb[4 * s + c];
      }
    } else {
#pragma unroll
      for (int s = 0; s < 16; ++s) {
        int idx = tid + 128 * s;
        Bs[buf][idx & 63][idx >> 6] = (_Float16)vb[s];
      }
    }
  };

  // ---- WMMA compute on LDS buffer `buf` -----------------------------------
  auto compute = [&](int buf) {
#pragma unroll
    for (int mi = 0; mi < 2; ++mi) {
      // A frag 16x32 f16: lane holds row (lane&15); elem j -> K = 16*(j/8)+8*h+j%8
      v16h a;
      const _Float16* Ap = &As[buf][wm * 32 + mi * 16 + r][0];
#pragma unroll
      for (int j = 0; j < 16; ++j)
        a[j] = Ap[((j >> 3) << 4) + (h << 3) + (j & 7)];

#pragma unroll
      for (int ni = 0; ni < 2; ++ni) {
        // B frag 32x16 f16: lane holds col (lane&15); elem j -> K = 16*(j/8)+8*h+j%8
        v16h b;
        const _Float16* Bp = &Bs[buf][wn * 32 + ni * 16 + r][h * 8];
#pragma unroll
        for (int j = 0; j < 16; ++j) b[j] = Bp[((j >> 3) << 4) + (j & 7)];

        acc[mi][ni] = __builtin_amdgcn_wmma_f32_16x16x32_f16(
            /*neg_a=*/false, a, /*neg_b=*/false, b,
            /*c_mod=*/(short)0, acc[mi][ni],
            /*reuse_a=*/false, /*reuse_b=*/false);
      }
    }
  };

  const int ntiles = (K + TK - 1) / TK;
  auto is_edge = [&](int t) { return !(fullMN && ((t + 1) * TK <= K)); };

  // prologue: stage tile 0
  load_tiles(0);
  store_tiles(0, is_edge(0));
  __syncthreads();

  for (int t = 0; t < ntiles; ++t) {
    int cur = t & 1;
    bool more = (t + 1 < ntiles);
    if (more) load_tiles((t + 1) * TK);   // global loads in flight during compute
    compute(cur);
    if (more) store_tiles(cur ^ 1, is_edge(t + 1));
    __syncthreads();                      // single barrier per k-tile
  }

  // C/D layout: VGPR v -> row = v + 8*h, col = lane&15
  if (fullMN) {
#pragma unroll
    for (int mi = 0; mi < 2; ++mi)
#pragma unroll
      for (int ni = 0; ni < 2; ++ni) {
        int col = bn + wn * 32 + ni * 16 + r;
        float bv = (flags & 1) ? bias[col] : 0.0f;
#pragma unroll
        for (int v = 0; v < 8; ++v) {
          int row = bm + wm * 32 + mi * 16 + h * 8 + v;
          float o = acc[mi][ni][v] + bv;
          if (flags & 2) o = fmaxf(o, 0.0f);
          C[(size_t)row * N + col] = o;
        }
      }
  } else {
#pragma unroll
    for (int mi = 0; mi < 2; ++mi)
#pragma unroll
      for (int ni = 0; ni < 2; ++ni) {
        int col = bn + wn * 32 + ni * 16 + r;
#pragma unroll
        for (int v = 0; v < 8; ++v) {
          int row = bm + wm * 32 + mi * 16 + h * 8 + v;
          if (row < M && col < N) {
            float o = acc[mi][ni][v];
            if (flags & 1) o += bias[col];
            if (flags & 2) o = fmaxf(o, 0.0f);
            C[(size_t)row * N + col] = o;
          }
        }
      }
  }
}

// ---------------- top-k pooling ---------------------------------------------

__global__ __launch_bounds__(128) void wnorm_inv(const float* __restrict__ w, int c,
                                                 float* __restrict__ scale) {
  __shared__ float red[128];
  int tid = threadIdx.x;
  float s = 0.0f;
  for (int i = tid; i < c; i += 128) s += w[i] * w[i];
  red[tid] = s;
  __syncthreads();
  for (int st = 64; st > 0; st >>= 1) {
    if (tid < st) red[tid] += red[tid + st];
    __syncthreads();
  }
  if (tid == 0) scale[0] = rsqrtf(red[0]);
}

__global__ __launch_bounds__(256) void node_scores(const float* __restrict__ x,
                                                   const float* __restrict__ w,
                                                   const float* __restrict__ scale,
                                                   float* __restrict__ score, int n, int c) {
  int i = blockIdx.x * blockDim.x + threadIdx.x;
  if (i < n) {
    float s = 0.0f;
    const float* xr = x + (size_t)i * c;
    for (int t = 0; t < c; ++t) s += xr[t] * w[t];
    score[i] = tanhf(s * scale[0]);
  }
}

// Single-block bitonic sort, descending (score, then ascending index for ties).
__global__ __launch_bounds__(1024) void topk_sort(const float* __restrict__ score,
                                                  int n, int k, int SZ,
                                                  int* __restrict__ perm,
                                                  float* __restrict__ vals) {
  __shared__ float skey[4096];
  __shared__ int   sidx[4096];
  int tid = threadIdx.x;
  for (int i = tid; i < SZ; i += 1024) {
    skey[i] = (i < n) ? score[i] : -3.0e38f;
    sidx[i] = i;
  }
  __syncthreads();
  for (int k2 = 2; k2 <= SZ; k2 <<= 1) {
    for (int j = k2 >> 1; j > 0; j >>= 1) {
      for (int i = tid; i < SZ; i += 1024) {
        int ixj = i ^ j;
        if (ixj > i) {
          float ka = skey[i], kb = skey[ixj];
          int ia = sidx[i], ib = sidx[ixj];
          bool aFirst = (ka > kb) || (ka == kb && ia < ib);
          bool descBlock = ((i & k2) == 0);
          if (descBlock ? !aFirst : aFirst) {
            skey[i] = kb; skey[ixj] = ka;
            sidx[i] = ib; sidx[ixj] = ia;
          }
        }
      }
      __syncthreads();
    }
  }
  for (int i = tid; i < k; i += 1024) {
    perm[i] = sidx[i];
    vals[i] = skey[i];
  }
}

__global__ __launch_bounds__(256) void gather_pool_x(const float* __restrict__ x,
                                                     const int* __restrict__ perm,
                                                     const float* __restrict__ vals,
                                                     float* __restrict__ xp, int k, int c) {
  int t = blockIdx.x * blockDim.x + threadIdx.x;
  if (t < k * c) {
    int i = t / c, j = t - i * c;
    xp[t] = x[(size_t)perm[i] * c + j] * vals[i];
  }
}

__global__ __launch_bounds__(256) void gather_pool_M(const float* __restrict__ B,
                                                     const int* __restrict__ perm,
                                                     float* __restrict__ Mp, int k, int n) {
  int t = blockIdx.x * blockDim.x + threadIdx.x;
  if (t < k * k) {
    int i = t / k, j = t - i * k;
    Mp[t] = B[(size_t)perm[i] * n + perm[j]];
  }
}

// out[perm[r], :] += xs[r, :]  (perm entries unique -> no atomics)
__global__ __launch_bounds__(256) void scatter_add_rows(float* __restrict__ out,
                                                        const float* __restrict__ xs,
                                                        const int* __restrict__ perm,
                                                        int k, int c) {
  int t = blockIdx.x * blockDim.x + threadIdx.x;
  if (t < k * c) {
    int r = t / c, j = t - r * c;
    out[(size_t)perm[r] * c + j] += xs[t];
  }
}

// ---------------------------------------------------------------------------

extern "C" void kernel_launch(void* const* d_in, const int* in_sizes, int n_in,
                              void* d_out, int out_size, void* d_ws, size_t ws_size,
                              hipStream_t stream) {
  const int CIN = 64, CH = 128, COUT = 64;
  const int N0 = in_sizes[0] / CIN;       // 3000
  const int E  = in_sizes[1] / 2;         // 48000
  const int ns[4] = {N0, (N0 + 1) / 2, (N0 + 3) / 4, (N0 + 7) / 8};  // 3000,1500,750,375

  const float* x_in = (const float*)d_in[0];
  const int*   ei   = (const int*)d_in[1];
  const float* dw[4] = {(const float*)d_in[2], (const float*)d_in[4],
                        (const float*)d_in[6], (const float*)d_in[8]};
  const float* db[4] = {(const float*)d_in[3], (const float*)d_in[5],
                        (const float*)d_in[7], (const float*)d_in[9]};
  const float* pw[3] = {(const float*)d_in[10], (const float*)d_in[11],
                        (const float*)d_in[12]};
  const float* uw[3] = {(const float*)d_in[13], (const float*)d_in[15],
                        (const float*)d_in[17]};
  const float* ub[3] = {(const float*)d_in[14], (const float*)d_in[16],
                        (const float*)d_in[18]};

  // ---- workspace carve-out (floats, 16B-aligned chunks) ----
  size_t off = 0;
  float* base = (float*)d_ws;
  auto allocF = [&](size_t cnt) {
    cnt = (cnt + 3) & ~(size_t)3;         // keep every chunk 16B aligned
    float* p = base + off; off += cnt; return p;
  };
  float* Madj[4];
  Madj[0] = allocF((size_t)ns[0] * ns[0]);
  Madj[1] = allocF((size_t)ns[1] * ns[1]);
  Madj[2] = allocF((size_t)ns[2] * ns[2]);
  Madj[3] = allocF((size_t)ns[3] * ns[3]);
  float* SCR1 = allocF((size_t)ns[0] * ns[0]);   // Ma / MnT scratch
  float* SCR2 = allocF((size_t)ns[0] * ns[0]);   // augmented adjacency B
  float* xs[4];
  xs[0] = allocF((size_t)ns[0] * CH);
  xs[1] = allocF((size_t)ns[1] * CH);
  xs[2] = allocF((size_t)ns[2] * CH);
  xs[3] = allocF((size_t)ns[3] * CH);
  float* Hbuf = allocF((size_t)ns[0] * CH);
  float* xa   = allocF((size_t)ns[0] * CH);
  float* xb   = allocF((size_t)ns[0] * CH);
  float* dinv  = allocF(ns[0]);
  float* score = allocF(ns[0]);
  float* vals  = allocF(ns[1]);
  float* scale = allocF(16);
  int* perm[3];
  perm[0] = (int*)allocF(ns[1]);
  perm[1] = (int*)allocF(ns[2]);
  perm[2] = (int*)allocF(ns[3]);

  auto cdiv = [](int a, int b) { return (a + b - 1) / b; };
  auto npow2 = [](int v) { int p = 1; while (p < v) p <<= 1; return p; };

  auto gemm = [&](const float* A, const float* B, float* C, const float* bias,
                  int M_, int N_, int K_, int flags) {
    dim3 g(cdiv(N_, TN), cdiv(M_, TM));
    gemm_wmma<<<g, 128, 0, stream>>>(A, B, C, bias, M_, N_, K_, flags);
  };

  // GCN: xout = act( MnT @ (xin @ W) + b ),  MnT materialized into SCR1.
  auto gcn = [&](const float* Ml, int n_, const float* xin, int cin,
                 const float* Wm, const float* bias, int cout, float* xout, bool relu) {
    col_degree_dinv<<<cdiv(n_, 256), 256, 0, stream>>>(Ml, dinv, n_);
    make_mnT<<<cdiv(n_ * n_, 256), 256, 0, stream>>>(Ml, dinv, SCR1, n_);
    gemm(xin, Wm, Hbuf, nullptr, n_, cout, cin, 0);
    gemm(SCR1, Hbuf, xout, bias, n_, cout, n_, 1 | (relu ? 2 : 0));
  };

  // ---- build dense adjacency M0 (no self loops) ----
  fill_f32<<<cdiv(ns[0] * ns[0], 256), 256, 0, stream>>>(Madj[0], 0.0f, ns[0] * ns[0]);
  build_adj<<<cdiv(E, 256), 256, 0, stream>>>(ei, Madj[0], E, ns[0]);
  clear_diag<<<cdiv(ns[0], 256), 256, 0, stream>>>(Madj[0], ns[0]);

  // ---- encoder ----
  gcn(Madj[0], ns[0], x_in, CIN, dw[0], db[0], CH, xs[0], true);

  for (int i = 0; i < 3; ++i) {
    int n = ns[i], k = ns[i + 1];
    // B = augment(M): (M offdiag + I)^2, zero diag
    prep_aug<<<cdiv(n * n, 256), 256, 0, stream>>>(Madj[i], SCR1, n);
    gemm(SCR1, SCR1, SCR2, nullptr, n, n, n, 0);
    clear_diag<<<cdiv(n, 256), 256, 0, stream>>>(SCR2, n);
    // top-k pool on B
    wnorm_inv<<<1, 128, 0, stream>>>(pw[i], CH, scale);
    node_scores<<<cdiv(n, 256), 256, 0, stream>>>(xs[i], pw[i], scale, score, n, CH);
    topk_sort<<<1, 1024, 0, stream>>>(score, n, k, npow2(n), perm[i], vals);
    gather_pool_x<<<cdiv(k * CH, 256), 256, 0, stream>>>(xs[i], perm[i], vals, xa, k, CH);
    gather_pool_M<<<cdiv(k * k, 256), 256, 0, stream>>>(SCR2, perm[i], Madj[i + 1], k, n);
    // GCN on pooled graph
    gcn(Madj[i + 1], k, xa, CH, dw[i + 1], db[i + 1], CH, xs[i + 1], true);
  }

  // ---- decoder (unpool + GCN, sum residual) ----
  // i=0: res=xs[2](750), perm[2], small=xs[3](375)
  copy_f32<<<cdiv(ns[2] * CH, 256), 256, 0, stream>>>(xa, xs[2], ns[2] * CH);
  scatter_add_rows<<<cdiv(ns[3] * CH, 256), 256, 0, stream>>>(xa, xs[3], perm[2], ns[3], CH);
  gcn(Madj[2], ns[2], xa, CH, uw[0], ub[0], CH, xb, true);

  // i=1: res=xs[1](1500), perm[1], small=xb(750)
  copy_f32<<<cdiv(ns[1] * CH, 256), 256, 0, stream>>>(xa, xs[1], ns[1] * CH);
  scatter_add_rows<<<cdiv(ns[2] * CH, 256), 256, 0, stream>>>(xa, xb, perm[1], ns[2], CH);
  gcn(Madj[1], ns[1], xa, CH, uw[1], ub[1], CH, xb, true);

  // i=2: res=xs[0](3000), perm[0], small=xb(1500) -> final 3000x64 output, no relu
  copy_f32<<<cdiv(ns[0] * CH, 256), 256, 0, stream>>>(xa, xs[0], ns[0] * CH);
  scatter_add_rows<<<cdiv(ns[1] * CH, 256), 256, 0, stream>>>(xa, xb, perm[0], ns[1], CH);
  gcn(Madj[0], ns[0], xa, CH, uw[2], ub[2], COUT, (float*)d_out, false);

  (void)n_in; (void)out_size; (void)ws_size;
}
